// MultiHeadAttention_47742856463003
// MI455X (gfx1250) — matrix operand order repaired and checked by reading, not compiled, not run on hardware
//
#include <hip/hip_runtime.h>
#include <hip/hip_bf16.h>

// MHA: B=2, T=2048, C=1024, H=16, D=64
#define BB 2
#define TT 2048
#define CC 1024
#define HH 16
#define DD 64
#define MM (BB*TT)        // 4096 rows
#define NQKV (3*CC)       // 3072

typedef _Float16 h8  __attribute__((ext_vector_type(8)));
typedef _Float16 h16 __attribute__((ext_vector_type(16)));
typedef float    f8  __attribute__((ext_vector_type(8)));

union H16U { h16 v; h8 h[2]; };

// Load a 16xf16 fragment as two 16-byte chunks (=> global_load_b128 x2 / ds_load_b128 x2)
template <typename P>
static __device__ __forceinline__ h16 ld2(P p0, P p1) {
  H16U u;
  u.h[0] = *(const h8*)p0;
  u.h[1] = *(const h8*)p1;
  return u.v;
}

static __device__ __forceinline__ f8 wmma_f16(h16 a, h16 b, f8 c) {
  return __builtin_amdgcn_wmma_f32_16x16x32_f16(false, a, false, b, (short)0, c, false, false);
}

// ---------------- precision / layout prep ----------------

__global__ void cvt_f32_f16(const float* __restrict__ in, _Float16* __restrict__ out, int n) {
  int i = blockIdx.x * blockDim.x + threadIdx.x;
  if (i < n) out[i] = (_Float16)in[i];
}

// out[n*K + k] = in[k*N + n]  (weights -> [N][K] f16, K-contiguous for B-matrix lanes)
__global__ void cvt_transpose(const float* __restrict__ in, _Float16* __restrict__ out, int K, int N) {
  int i = blockIdx.x * blockDim.x + threadIdx.x;
  if (i < K * N) {
    int n = i / K, k = i - n * K;
    out[i] = (_Float16)in[k * N + n];
  }
}

// ---------------- QKV projection:  xh[4096,1024] @ wT[3072,1024]^T + b ----------------
// wave = 16 rows x 64 cols, software-pipelined (prefetch k+32 while WMMA on k).
// Writes Q,K as [B,H,T,D] f16, V transposed as [B,H,D,T] f16.

__global__ __launch_bounds__(128) void gemm_qkv(const _Float16* __restrict__ xh,
                                                const _Float16* __restrict__ wT,
                                                const float* __restrict__ bias,
                                                _Float16* __restrict__ qb,
                                                _Float16* __restrict__ kb,
                                                _Float16* __restrict__ vt) {
  const int lane = threadIdx.x & 31;
  const int wid  = threadIdx.x >> 5;
  const int ln = lane & 15;
  const int hf = lane >> 4;
  const int m0 = blockIdx.y * 64 + wid * 16;
  const int n0 = blockIdx.x * 64;

  const _Float16* arow = xh + (m0 + ln) * CC + hf * 16;
  const _Float16* b0p = wT + (n0 +  0 + ln) * CC + hf * 16;
  const _Float16* b1p = wT + (n0 + 16 + ln) * CC + hf * 16;
  const _Float16* b2p = wT + (n0 + 32 + ln) * CC + hf * 16;
  const _Float16* b3p = wT + (n0 + 48 + ln) * CC + hf * 16;

  h16 a  = ld2(arow, arow + 8);
  h16 b0 = ld2(b0p, b0p + 8);
  h16 b1 = ld2(b1p, b1p + 8);
  h16 b2 = ld2(b2p, b2p + 8);
  h16 b3 = ld2(b3p, b3p + 8);

  f8 acc[4] = {};
  for (int k0 = 0; k0 < CC; k0 += 32) {
    const int kn = (k0 + 32 < CC) ? (k0 + 32) : 0;   // branch-free tail: reload step 0
    h16 an  = ld2(arow + kn, arow + kn + 8);
    h16 bn0 = ld2(b0p + kn, b0p + kn + 8);
    h16 bn1 = ld2(b1p + kn, b1p + kn + 8);
    h16 bn2 = ld2(b2p + kn, b2p + kn + 8);
    h16 bn3 = ld2(b3p + kn, b3p + kn + 8);

    acc[0] = wmma_f16(a, b0, acc[0]);
    acc[1] = wmma_f16(a, b1, acc[1]);
    acc[2] = wmma_f16(a, b2, acc[2]);
    acc[3] = wmma_f16(a, b3, acc[3]);

    a = an; b0 = bn0; b1 = bn1; b2 = bn2; b3 = bn3;
  }

#pragma unroll
  for (int s = 0; s < 4; ++s) {
    const int ncol = n0 + s * 16 + ln;          // 0..3071
    const float bv = bias[ncol];
    const int part = ncol >> 10;                // 0=Q 1=K 2=V
    const int c = ncol & (CC - 1);
    const int h = c >> 6, d = c & 63;
#pragma unroll
    for (int r = 0; r < 8; ++r) {
      const int mrow = m0 + r + 8 * hf;
      const int b_ = mrow >> 11, t = mrow & (TT - 1);
      const _Float16 val = (_Float16)(acc[s][r] + bv);
      if (part == 0)      qb[(((b_ * HH + h) * TT) + t) * DD + d] = val;
      else if (part == 1) kb[(((b_ * HH + h) * TT) + t) * DD + d] = val;
      else                vt[(((b_ * HH + h) * DD) + d) * TT + t] = val;
    }
  }
}

// ---------------- flash attention, causal, one wave per 16-query tile ----------------
// K-tile fragments double-buffered; V-tile loads issued before the softmax VALU work
// so VMEM latency hides under exp/shuffle/LDS traffic.

__global__ __launch_bounds__(128) void attn_fwd(const _Float16* __restrict__ qb,
                                                const _Float16* __restrict__ kb,
                                                const _Float16* __restrict__ vt,
                                                _Float16* __restrict__ att) {
  __shared__ __align__(16) _Float16 lds[4][16 * 32];
  const int lane = threadIdx.x & 31;
  const int wid  = threadIdx.x >> 5;
  const int ln = lane & 15;
  const int hf = lane >> 4;

  const int bh = blockIdx.y;                 // b*16 + h
  const int qt = blockIdx.x + 32 * wid;      // 0..127
  const int q0 = qt * 16;

  const _Float16* Q  = qb + (size_t)bh * TT * DD;
  const _Float16* Km = kb + (size_t)bh * TT * DD;
  const _Float16* Vt = vt + (size_t)bh * DD * TT;
  _Float16* ldsw = &lds[wid][0];

  // Q tile kept in registers for the whole key loop (A-matrix layout, D=64 -> 2 chunks)
  const _Float16* qrow = Q + (q0 + ln) * DD + hf * 16;
  const h16 aq0 = ld2(qrow,      qrow + 8);    // d 0..31
  const h16 aq1 = ld2(qrow + 32, qrow + 40);   // d 32..63

  f8 o[4] = {};
  float mrun[8], lrun[8];
#pragma unroll
  for (int r = 0; r < 8; ++r) { mrun[r] = -1e30f; lrun[r] = 0.0f; }

  const int kend = ((q0 + 16 + 31) >> 5) << 5; // causal: keys < kend

  // preload K-tile fragments for kk = 0
  const _Float16* kbase = Km + ln * DD + hf * 16;
  h16 kb00 = ld2(kbase,                 kbase + 8);
  h16 kb01 = ld2(kbase + 32,            kbase + 40);
  h16 kb10 = ld2(kbase + 16 * DD,       kbase + 16 * DD + 8);
  h16 kb11 = ld2(kbase + 16 * DD + 32,  kbase + 16 * DD + 40);

  for (int kk = 0; kk < kend; kk += 32) {
    // ---- scores: two 16x16 subtiles, contraction over D=64 ----
    f8 s0 = {}, s1 = {};
    s0 = wmma_f16(aq0, kb00, s0);
    s0 = wmma_f16(aq1, kb01, s0);
    s1 = wmma_f16(aq0, kb10, s1);
    s1 = wmma_f16(aq1, kb11, s1);

    // ---- prefetch V fragments for THIS step (consumed after softmax) ----
    h16 vb[4];
#pragma unroll
    for (int s = 0; s < 4; ++s) {
      const _Float16* vp = Vt + (s * 16 + ln) * TT + kk + hf * 16;
      vb[s] = ld2(vp, vp + 8);
    }

    // ---- prefetch K fragments for NEXT step (in flight under softmax + P*V) ----
    {
      const int kn = (kk + 32 < kend) ? (kk + 32) : 0;
      const _Float16* krn = kbase + kn * DD;
      kb00 = ld2(krn,                krn + 8);
      kb01 = ld2(krn + 32,           krn + 40);
      kb10 = ld2(krn + 16 * DD,      krn + 16 * DD + 8);
      kb11 = ld2(krn + 16 * DD + 32, krn + 16 * DD + 40);
    }

    // ---- scale + causal mask + online softmax (rows live in 16-lane halves) ----
#pragma unroll
    for (int r = 0; r < 8; ++r) {
      const int q = q0 + r + 8 * hf;
      const float v0 = (kk + ln      <= q) ? s0[r] * 0.125f : -1e30f;
      const float v1 = (kk + 16 + ln <= q) ? s1[r] * 0.125f : -1e30f;

      float m = fmaxf(v0, v1);
#pragma unroll
      for (int off = 8; off >= 1; off >>= 1) m = fmaxf(m, __shfl_xor(m, off, 32));
      const float mnew  = fmaxf(mrun[r], m);
      const float alpha = __expf(mrun[r] - mnew);
      const float p0 = __expf(v0 - mnew);
      const float p1 = __expf(v1 - mnew);
      float rs = p0 + p1;
#pragma unroll
      for (int off = 8; off >= 1; off >>= 1) rs += __shfl_xor(rs, off, 32);
      lrun[r] = lrun[r] * alpha + rs;
      mrun[r] = mnew;
#pragma unroll
      for (int s = 0; s < 4; ++s) o[s][r] = o[s][r] * alpha;

      const int row = r + 8 * hf;              // C-layout row owned by this lane group
      ldsw[row * 32 + ln]      = (_Float16)p0;
      ldsw[row * 32 + 16 + ln] = (_Float16)p1;
    }

    asm volatile("s_wait_dscnt 0" ::: "memory");   // P in LDS visible to own wave

    // reload P as A-matrix (16x32), multiply by V (transposed => contiguous-K B tiles)
    const _Float16* pr = ldsw + ln * 32 + hf * 16;
    const h16 ap = ld2(pr, pr + 8);
#pragma unroll
    for (int s = 0; s < 4; ++s) o[s] = wmma_f16(ap, vb[s], o[s]);
  }

  // ---- normalize and store to att[B,T,C] f16 (c = h*64 + d) ----
  const int b_ = bh >> 4;
  const int h  = bh & 15;
#pragma unroll
  for (int r = 0; r < 8; ++r) {
    const float inv = 1.0f / lrun[r];
    const int t = q0 + r + 8 * hf;
#pragma unroll
    for (int s = 0; s < 4; ++s) {
      const int d = s * 16 + ln;
      att[((size_t)(b_ * TT + t)) * CC + h * DD + d] = (_Float16)(o[s][r] * inv);
    }
  }
}

// ---------------- output projection: att[4096,1024] @ woutT[1024,1024]^T + b -> f32 ----------------

__global__ __launch_bounds__(128) void gemm_out(const _Float16* __restrict__ att,
                                                const _Float16* __restrict__ wT,
                                                const float* __restrict__ bias,
                                                float* __restrict__ out) {
  const int lane = threadIdx.x & 31;
  const int wid  = threadIdx.x >> 5;
  const int ln = lane & 15;
  const int hf = lane >> 4;
  const int m0 = blockIdx.y * 64 + wid * 16;
  const int n0 = blockIdx.x * 64;

  const _Float16* arow = att + (m0 + ln) * CC + hf * 16;
  const _Float16* b0p = wT + (n0 +  0 + ln) * CC + hf * 16;
  const _Float16* b1p = wT + (n0 + 16 + ln) * CC + hf * 16;
  const _Float16* b2p = wT + (n0 + 32 + ln) * CC + hf * 16;
  const _Float16* b3p = wT + (n0 + 48 + ln) * CC + hf * 16;

  h16 a  = ld2(arow, arow + 8);
  h16 b0 = ld2(b0p, b0p + 8);
  h16 b1 = ld2(b1p, b1p + 8);
  h16 b2 = ld2(b2p, b2p + 8);
  h16 b3 = ld2(b3p, b3p + 8);

  f8 acc[4] = {};
  for (int k0 = 0; k0 < CC; k0 += 32) {
    const int kn = (k0 + 32 < CC) ? (k0 + 32) : 0;
    h16 an  = ld2(arow + kn, arow + kn + 8);
    h16 bn0 = ld2(b0p + kn, b0p + kn + 8);
    h16 bn1 = ld2(b1p + kn, b1p + kn + 8);
    h16 bn2 = ld2(b2p + kn, b2p + kn + 8);
    h16 bn3 = ld2(b3p + kn, b3p + kn + 8);

    acc[0] = wmma_f16(a, b0, acc[0]);
    acc[1] = wmma_f16(a, b1, acc[1]);
    acc[2] = wmma_f16(a, b2, acc[2]);
    acc[3] = wmma_f16(a, b3, acc[3]);

    a = an; b0 = bn0; b1 = bn1; b2 = bn2; b3 = bn3;
  }

#pragma unroll
  for (int s = 0; s < 4; ++s) {
    const int ncol = n0 + s * 16 + ln;
    const float bv = bias[ncol];
#pragma unroll
    for (int r = 0; r < 8; ++r) {
      const int mrow = m0 + r + 8 * hf;
      out[(size_t)mrow * CC + ncol] = acc[s][r] + bv;
    }
  }
}

// ---------------- launch ----------------

extern "C" void kernel_launch(void* const* d_in, const int* in_sizes, int n_in,
                              void* d_out, int out_size, void* d_ws, size_t ws_size,
                              hipStream_t stream) {
  (void)in_sizes; (void)n_in; (void)out_size; (void)ws_size;
  const float* x     = (const float*)d_in[0];
  const float* w_qkv = (const float*)d_in[1];
  const float* b_qkv = (const float*)d_in[2];
  const float* w_out = (const float*)d_in[3];
  const float* b_out = (const float*)d_in[4];
  float* out = (float*)d_out;

  // workspace carve-up (f16 elements); total = 48 MiB
  _Float16* ws    = (_Float16*)d_ws;
  _Float16* xh    = ws;                         // 4096*1024
  _Float16* wqkvT = xh    + (size_t)MM * CC;    // 3072*1024
  _Float16* woutT = wqkvT + (size_t)NQKV * CC;  // 1024*1024
  _Float16* qb    = woutT + (size_t)CC * CC;    // [B,H,T,D]
  _Float16* kb    = qb    + (size_t)MM * CC;    // [B,H,T,D]
  _Float16* vt    = kb    + (size_t)MM * CC;    // [B,H,D,T]
  _Float16* att   = vt    + (size_t)MM * CC;    // [B,T,C]

  cvt_f32_f16 <<<(MM * CC) / 256, 256, 0, stream>>>(x, xh, MM * CC);
  cvt_transpose<<<(NQKV * CC) / 256, 256, 0, stream>>>(w_qkv, wqkvT, CC, NQKV);
  cvt_transpose<<<(CC * CC) / 256, 256, 0, stream>>>(w_out, woutT, CC, CC);

  gemm_qkv<<<dim3(NQKV / 64, MM / 64), 128, 0, stream>>>(xh, wqkvT, b_qkv, qb, kb, vt);
  attn_fwd<<<dim3(TT / 64, BB * HH), 128, 0, stream>>>(qb, kb, vt, att);
  gemm_out<<<dim3(CC / 64, MM / 64), 128, 0, stream>>>(att, woutT, b_out, out);
}
